// Decoder_RNN_48911087567585
// MI455X (gfx1250) — matrix operand order repaired and checked by reading, not compiled, not run on hardware
//
#include <hip/hip_runtime.h>
#include <hip/hip_bf16.h>
#include <math.h>

typedef __attribute__((ext_vector_type(16))) _Float16 v16h;
typedef __attribute__((ext_vector_type(8)))  float    v8f;

#define Bsz   64
#define Tlen  2048
#define Hdim  256
#define Edim  256
#define Vocab 33
#define Lsteps 200
#define MLPH  512
#define NCH   16            // attention T-chunks per batch
#define TC    (Tlen / NCH)  // 128 timesteps per chunk

// ---- workspace layout (floats) ----
#define H_OFF     32768      // [3][2][64][256]  ping-pong h state
#define C_OFF     131072     // [3][64][256]
#define MLP_OFF   196608     // [64][512]
#define CTXP_OFF  229376     // [64][16][256] partial contexts
#define MI_OFF    491520     // [64][16] chunk maxima
#define SI_OFF    492544     // [64][16] chunk exp-sums
#define F32_TOTAL 493568
// packed f16 weights start at byte F32_TOTAL*4
#define PW0_OFF  0           // 1024x768  -> 786432 halfs
#define PW1_OFF  786432      // 1024x512
#define PW2_OFF  1310720     // 1024x512
#define PFC1_OFF 1835008     // 512x512

__device__ __forceinline__ float sigf(float x) { return 1.0f / (1.0f + __expf(-x)); }

// inline flash-softmax combine: ctx[b][d..d+3] from per-chunk partials.
// At init (all zeros): S=0 -> acc/eps = 0, matching reference ctx0 == 0.
__device__ __forceinline__ float4 combine4(const float* __restrict__ Mi,
                                           const float* __restrict__ Si,
                                           const float* __restrict__ ctxp,
                                           int b, int d) {
    float M = -3.0e38f;
    #pragma unroll
    for (int c = 0; c < NCH; ++c) M = fmaxf(M, Mi[b * NCH + c]);
    float S = 0.0f;
    float4 acc = make_float4(0.f, 0.f, 0.f, 0.f);
    #pragma unroll
    for (int c = 0; c < NCH; ++c) {
        float w = __expf(Mi[b * NCH + c] - M);
        S += w * Si[b * NCH + c];
        float4 cp = *(const float4*)&ctxp[((size_t)(b * NCH + c)) * Hdim + d];
        acc.x += w * cp.x; acc.y += w * cp.y; acc.z += w * cp.z; acc.w += w * cp.w;
    }
    float sc = 1.0f / fmaxf(S, 1e-12f);
    acc.x *= sc; acc.y *= sc; acc.z *= sc; acc.w *= sc;
    return acc;
}

// ---------------- utility: zero ----------------
__global__ void zero_kernel(float* p, int n) {
    int i = blockIdx.x * blockDim.x + threadIdx.x;
    if (i < n) p[i] = 0.0f;
}

// ---------------- weight packing into WMMA B-fragment order ----------------
// B(k,n) = W[n,k];  W = [w_ih | w_hh] row n, concatenated over k.
__global__ void pack_weights(const float* __restrict__ w_ih,
                             const float* __restrict__ w_hh,
                             _Float16* __restrict__ dst,
                             int In, int Ktot, int N) {
    int i = blockIdx.x * blockDim.x + threadIdx.x;
    int nkt = Ktot >> 5;
    int total = N * Ktot;
    if (i >= total) return;
    int jt   = i / (nkt * 512);
    int rem  = i % (nkt * 512);
    int kt   = rem / 512;
    int lane = (rem % 512) / 16;
    int h    = i & 15;
    int n = jt * 16 + (lane & 15);
    int k = kt * 32 + (h >> 3) * 16 + (lane >> 4) * 8 + (h & 7);
    float v;
    if (k < In) v = w_ih[(size_t)n * In + k];
    else        v = w_hh[(size_t)n * Hdim + (k - In)];
    dst[i] = (_Float16)v;
}

// ============ shared GEMM machinery (16x16 tile per wave, f16 WMMA) =========
// A-fragment gather per ISA 16-bit A 16x32 layout; B pre-packed lane-contiguous.
#define GEMM_BODY(NKT, LOADCHUNK)                                              \
    storechunk(0, LOADCHUNK(0));                                               \
    __syncthreads();                                                           \
    v8f acc = {};                                                              \
    int p = 0;                                                                 \
    const int m = mt * 16 + (lane & 15);                                       \
    for (int kt = 0; kt < (NKT); ++kt) {                                       \
        float4 nv;                                                             \
        const bool have = (kt + 1 < (NKT));                                    \
        if (have) nv = LOADCHUNK(kt + 1);                                      \
        v16h a;                                                                \
        _Pragma("unroll")                                                      \
        for (int h = 0; h < 16; ++h) {                                         \
            int kl = ((h >> 3) * 16) + (h & 7) + hi * 8;                       \
            a[h] = aslab[p][m][kl];                                            \
        }                                                                      \
        const _Float16* bp = wpack + (((size_t)jt * (NKT) + kt) * 32 + lane) * 16; \
        v16h bf = *(const v16h*)bp;                                            \
        acc = __builtin_amdgcn_wmma_f32_16x16x32_f16(                          \
            false, a, false, bf, (short)0, acc, false, false);                 \
        if (have) storechunk(p ^ 1, nv);                                       \
        __syncthreads();                                                       \
        p ^= 1;                                                                \
    }

// -------- LSTM layer 0: A = [emb[labels[t]] | combine(ctx) | h0_prev] -------
// 24 k-chunks: [0,8)=emb, [8,16)=ctx, [16,24)=h.  block bh owns 16 hidden
// cols across all 4 gates; 16 waves: mt=w&3 (batch tile), g=w>>2 (gate).
__global__ __launch_bounds__(512)
void lstm_layer0(const int* __restrict__ labels, int t,
                 const float* __restrict__ emb,
                 const float* __restrict__ Mi, const float* __restrict__ Si,
                 const float* __restrict__ ctxp,
                 const float* __restrict__ hprev,
                 const _Float16* __restrict__ wpack,
                 const float* __restrict__ b_ih, const float* __restrict__ b_hh,
                 float* __restrict__ c_state, float* __restrict__ h_out) {
    const int bh = blockIdx.x, tid = threadIdx.x;
    const int wave = tid >> 5, lane = tid & 31;
    const int mt = wave & 3, g = wave >> 2;
    const int jt = g * 16 + bh;
    const int hi = lane >> 4;

    __shared__ _Float16 aslab[2][64][36];
    __shared__ float    gbuf[64][65];

    const int e4 = tid * 4;
    const int br = e4 >> 5;     // staged row 0..63
    const int kk = e4 & 31;     // 0,4,...,28
    const int lab = labels[t * Bsz + br];

    auto loadchunk = [&](int kt) -> float4 {
        int base = kt * 32;
        if (base < Edim)
            return *(const float4*)&emb[(size_t)lab * Edim + base + kk];
        else if (base < Edim + Hdim)
            return combine4(Mi, Si, ctxp, br, base - Edim + kk);
        else
            return *(const float4*)&hprev[br * Hdim + base - (Edim + Hdim) + kk];
    };
    auto storechunk = [&](int sl, float4 v) {
        aslab[sl][br][kk + 0] = (_Float16)v.x;
        aslab[sl][br][kk + 1] = (_Float16)v.y;
        aslab[sl][br][kk + 2] = (_Float16)v.z;
        aslab[sl][br][kk + 3] = (_Float16)v.w;
    };

    GEMM_BODY(24, loadchunk)

    int n = lane & 15;
    int col = g * Hdim + bh * 16 + n;
    float bias = b_ih[col] + b_hh[col];
    #pragma unroll
    for (int r = 0; r < 8; ++r)
        gbuf[mt * 16 + r + 8 * hi][g * 16 + n] = acc[r] + bias;
    __syncthreads();
    for (int e = tid; e < 64 * 16; e += 512) {
        int b = e >> 4, hc = e & 15;
        float gi = gbuf[b][hc], gf = gbuf[b][16 + hc];
        float gg = gbuf[b][32 + hc], go = gbuf[b][48 + hc];
        int ci = b * Hdim + bh * 16 + hc;
        float cp = c_state[ci];
        float c2 = sigf(gf) * cp + sigf(gi) * tanhf(gg);
        c_state[ci] = c2;
        h_out[ci] = sigf(go) * tanhf(c2);
    }
}

// -------- LSTM layers 1,2: A = [h_below_cur | h_self_prev] ------------------
__global__ __launch_bounds__(512)
void lstm_layer(const float* __restrict__ src0,
                const float* __restrict__ src1,
                const _Float16* __restrict__ wpack,
                const float* __restrict__ b_ih, const float* __restrict__ b_hh,
                float* __restrict__ c_state, float* __restrict__ h_out) {
    const int bh = blockIdx.x, tid = threadIdx.x;
    const int wave = tid >> 5, lane = tid & 31;
    const int mt = wave & 3, g = wave >> 2;
    const int jt = g * 16 + bh;
    const int hi = lane >> 4;

    __shared__ _Float16 aslab[2][64][36];
    __shared__ float    gbuf[64][65];

    const int e4 = tid * 4;
    const int br = e4 >> 5;
    const int kk = e4 & 31;

    auto loadchunk = [&](int kt) -> float4 {
        int base = kt * 32;
        const float* s = (base < Hdim) ? (src0 + (size_t)br * Hdim + base + kk)
                                       : (src1 + (size_t)br * Hdim + base - Hdim + kk);
        return *(const float4*)s;
    };
    auto storechunk = [&](int sl, float4 v) {
        aslab[sl][br][kk + 0] = (_Float16)v.x;
        aslab[sl][br][kk + 1] = (_Float16)v.y;
        aslab[sl][br][kk + 2] = (_Float16)v.z;
        aslab[sl][br][kk + 3] = (_Float16)v.w;
    };

    GEMM_BODY(16, loadchunk)

    int n = lane & 15;
    int col = g * Hdim + bh * 16 + n;
    float bias = b_ih[col] + b_hh[col];
    #pragma unroll
    for (int r = 0; r < 8; ++r)
        gbuf[mt * 16 + r + 8 * hi][g * 16 + n] = acc[r] + bias;
    __syncthreads();
    for (int e = tid; e < 64 * 16; e += 512) {
        int b = e >> 4, hc = e & 15;
        float gi = gbuf[b][hc], gf = gbuf[b][16 + hc];
        float gg = gbuf[b][32 + hc], go = gbuf[b][48 + hc];
        int ci = b * Hdim + bh * 16 + hc;
        float cp = c_state[ci];
        float c2 = sigf(gf) * cp + sigf(gi) * tanhf(gg);
        c_state[ci] = c2;
        h_out[ci] = sigf(go) * tanhf(c2);
    }
}

// ---------------- attention pass 1: per-chunk partial flash softmax ---------
__global__ __launch_bounds__(256)
void attn_partial(const float* __restrict__ key,
                  const float* __restrict__ value,
                  const int* __restrict__ seqlens,
                  const float* __restrict__ rnn_out,
                  float* __restrict__ Mi, float* __restrict__ Si,
                  float* __restrict__ ctxp) {
    const int b  = blockIdx.x >> 4;
    const int ch = blockIdx.x & 15;
    const int t0 = ch * TC;
    const int tid = threadIdx.x;
    const int wave = tid >> 5, lane = tid & 31;

    __shared__ float rs[Hdim];
    __shared__ float es[TC];
    __shared__ float red1[8], red2[8];

    rs[tid] = rnn_out[b * Hdim + tid];
    __syncthreads();

    for (int tt = wave; tt < TC; tt += 8) {
        const float* kp = key + ((size_t)b * Tlen + t0 + tt) * Hdim;
        if (tt + 8 < TC)
            __builtin_prefetch(key + ((size_t)b * Tlen + t0 + tt + 8) * Hdim + lane * 4, 0, 0);
        float p = 0.0f;
        #pragma unroll
        for (int d = lane; d < Hdim; d += 32) p += kp[d] * rs[d];
        #pragma unroll
        for (int off = 16; off; off >>= 1) p += __shfl_xor(p, off, 32);
        if (lane == 0) es[tt] = p;
    }
    __syncthreads();

    float mx = -3.0e38f;
    for (int tt = tid; tt < TC; tt += 256) mx = fmaxf(mx, es[tt]);
    #pragma unroll
    for (int off = 16; off; off >>= 1) mx = fmaxf(mx, __shfl_xor(mx, off, 32));
    if (lane == 0) red1[wave] = mx;
    __syncthreads();
    float MX = red1[0];
    #pragma unroll
    for (int w = 1; w < 8; ++w) MX = fmaxf(MX, red1[w]);

    const int slen = seqlens[b];
    float sum = 0.0f;
    for (int tt = tid; tt < TC; tt += 256) {
        float pe = ((t0 + tt) < slen) ? __expf(es[tt] - MX) : 0.0f;
        es[tt] = pe;
        sum += pe;
    }
    #pragma unroll
    for (int off = 16; off; off >>= 1) sum += __shfl_xor(sum, off, 32);
    if (lane == 0) red2[wave] = sum;
    __syncthreads();
    float S = 0.0f;
    #pragma unroll
    for (int w = 0; w < 8; ++w) S += red2[w];
    if (tid == 0) { Mi[b * NCH + ch] = MX; Si[b * NCH + ch] = S; }

    float acc = 0.0f;
    const float* vp = value + ((size_t)b * Tlen + t0) * Hdim;
    #pragma unroll 8
    for (int tt = 0; tt < TC; ++tt) acc += es[tt] * vp[(size_t)tt * Hdim + tid];
    ctxp[((size_t)(b * NCH + ch)) * Hdim + tid] = acc;
}

// ---------------- fc1: [64 x 512] x [512 x 512] + bias + ReLU (WMMA) -------
// A = [rnn_out | combine(ctx)]; flash combine inlined in staging.
__global__ __launch_bounds__(512)
void fc1_kernel(const float* __restrict__ src0,   // rnn_out [64,256]
                const float* __restrict__ Mi, const float* __restrict__ Si,
                const float* __restrict__ ctxp,
                const _Float16* __restrict__ wpack,
                const float* __restrict__ bias,
                float* __restrict__ out) {
    const int tid = threadIdx.x;
    const int wave = tid >> 5, lane = tid & 31;
    const int mt = wave & 3;
    const int jt = blockIdx.x * 4 + (wave >> 2);  // grid = 8 -> 32 n-tiles
    const int hi = lane >> 4;

    __shared__ _Float16 aslab[2][64][36];

    const int e4 = tid * 4;
    const int br = e4 >> 5;
    const int kk = e4 & 31;

    auto loadchunk = [&](int kt) -> float4 {
        int base = kt * 32;
        if (base < Hdim)
            return *(const float4*)&src0[(size_t)br * Hdim + base + kk];
        else
            return combine4(Mi, Si, ctxp, br, base - Hdim + kk);
    };
    auto storechunk = [&](int sl, float4 v) {
        aslab[sl][br][kk + 0] = (_Float16)v.x;
        aslab[sl][br][kk + 1] = (_Float16)v.y;
        aslab[sl][br][kk + 2] = (_Float16)v.z;
        aslab[sl][br][kk + 3] = (_Float16)v.w;
    };

    GEMM_BODY(16, loadchunk)

    int n = lane & 15;
    int col = jt * 16 + n;
    float bv = bias[col];
    #pragma unroll
    for (int r = 0; r < 8; ++r) {
        int row = mt * 16 + r + 8 * hi;
        out[row * MLPH + col] = fmaxf(acc[r] + bv, 0.0f);
    }
}

// ---------------- fc2: [64 x 512] x [512 x 33] + bias -> y_hat[b,t,:] ------
__global__ void fc2_kernel(const float* __restrict__ mlp,
                           const float* __restrict__ w,
                           const float* __restrict__ bias,
                           float* __restrict__ out, int t) {
    int b = blockIdx.x, v = threadIdx.x;
    if (v >= Vocab) return;
    const float* row = mlp + (size_t)b * MLPH;
    const float* wr  = w + (size_t)v * MLPH;
    float acc = bias[v];
    #pragma unroll 8
    for (int k = 0; k < MLPH; ++k) acc += row[k] * wr[k];
    out[((size_t)b * Lsteps + t) * Vocab + v] = acc;
}

extern "C" void kernel_launch(void* const* d_in, const int* in_sizes, int n_in,
                              void* d_out, int out_size, void* d_ws, size_t ws_size,
                              hipStream_t stream) {
    (void)in_sizes; (void)n_in; (void)out_size; (void)ws_size;
    const float* key     = (const float*)d_in[0];
    const float* value   = (const float*)d_in[1];
    const int*   labels  = (const int*)  d_in[2];
    const int*   seqlens = (const int*)  d_in[3];
    const float* emb     = (const float*)d_in[4];
    const float* w_ih0   = (const float*)d_in[5];
    const float* w_hh0   = (const float*)d_in[6];
    const float* b_ih0   = (const float*)d_in[7];
    const float* b_hh0   = (const float*)d_in[8];
    const float* w_ih_r  = (const float*)d_in[9];
    const float* w_hh_r  = (const float*)d_in[10];
    const float* b_ih_r  = (const float*)d_in[11];
    const float* b_hh_r  = (const float*)d_in[12];
    const float* fc1_w   = (const float*)d_in[13];
    const float* fc1_b   = (const float*)d_in[14];
    const float* fc2_w   = (const float*)d_in[15];
    const float* fc2_b   = (const float*)d_in[16];
    float* out = (float*)d_out;

    float*     wsf = (float*)d_ws;
    _Float16*  wsh = (_Float16*)((char*)d_ws + (size_t)F32_TOTAL * 4);

    float* mlp  = wsf + MLP_OFF;
    float* ctxp = wsf + CTXP_OFF;
    float* Mi   = wsf + MI_OFF;
    float* Si   = wsf + SI_OFF;
    auto hbuf = [&](int l, int p) { return wsf + H_OFF + ((size_t)(l * 2 + p)) * (Bsz * Hdim); };
    auto cbuf = [&](int l)        { return wsf + C_OFF + (size_t)l * (Bsz * Hdim); };
    _Float16* pw0  = wsh + PW0_OFF;
    _Float16* pw1  = wsh + PW1_OFF;
    _Float16* pw2  = wsh + PW2_OFF;
    _Float16* pfc1 = wsh + PFC1_OFF;

    // zero all f32 state (h, c, Mi/Si/ctxp must start 0 so step-0 combine == 0)
    zero_kernel<<<(F32_TOTAL + 255) / 256, 256, 0, stream>>>(wsf, F32_TOTAL);

    // pack weights once (concatenated [W_ih | W_hh], B-fragment order)
    pack_weights<<<(1024 * 768 + 255) / 256, 256, 0, stream>>>(w_ih0, w_hh0, pw0, 512, 768, 1024);
    pack_weights<<<(1024 * 512 + 255) / 256, 256, 0, stream>>>(w_ih_r, w_hh_r, pw1, 256, 512, 1024);
    pack_weights<<<(1024 * 512 + 255) / 256, 256, 0, stream>>>(w_ih_r + (size_t)1024 * 256,
                                                               w_hh_r + (size_t)1024 * 256,
                                                               pw2, 256, 512, 1024);
    pack_weights<<<(512 * 512 + 255) / 256, 256, 0, stream>>>(fc1_w, nullptr, pfc1, 512, 512, 512);

    for (int t = 0; t < Lsteps; ++t) {
        int pp = t & 1;        // prev-h parity
        int np = pp ^ 1;       // new-h parity
        lstm_layer0<<<16, 512, 0, stream>>>(labels, t, emb, Mi, Si, ctxp, hbuf(0, pp),
                                            pw0, b_ih0, b_hh0, cbuf(0), hbuf(0, np));
        lstm_layer<<<16, 512, 0, stream>>>(hbuf(0, np), hbuf(1, pp), pw1,
                                           b_ih_r, b_hh_r, cbuf(1), hbuf(1, np));
        lstm_layer<<<16, 512, 0, stream>>>(hbuf(1, np), hbuf(2, pp), pw2,
                                           b_ih_r + 1024, b_hh_r + 1024, cbuf(2), hbuf(2, np));
        attn_partial<<<Bsz * NCH, 256, 0, stream>>>(key, value, seqlens, hbuf(2, np),
                                                    Mi, Si, ctxp);
        fc1_kernel<<<8, 512, 0, stream>>>(hbuf(2, np), Mi, Si, ctxp, pfc1, fc1_b, mlp);
        fc2_kernel<<<Bsz, 64, 0, stream>>>(mlp, fc2_w, fc2_b, out, t);
    }
}
